// FlashMixtralLayer_78331613545179
// MI455X (gfx1250) — matrix operand order repaired and checked by reading, not compiled, not run on hardware
//
#include <hip/hip_runtime.h>
#include <hip/hip_bf16.h>
#include <math.h>

// ---------------- problem constants ----------------
static constexpr int kB   = 4;
static constexpr int kS   = 2048;
static constexpr int kH   = 1024;
static constexpr int kF   = 3584;     // FFN
static constexpr int kE   = 8;
static constexpr int kT   = kB * kS;  // 8192 tokens
static constexpr int kTopK = 2;

static constexpr int BM = 128;        // rows (slots) per block; expert segments padded to BM
static constexpr int ROWS_CAP = kTopK * kT + kE * BM;  // 16384 + 1024 = 17408

typedef unsigned short u16;
typedef __bf16 bf16_t;
typedef bf16_t bf16x16 __attribute__((ext_vector_type(16)));
typedef float  f32x8   __attribute__((ext_vector_type(8)));

union FragU { uint4 u[2]; bf16x16 v; };

__device__ __forceinline__ u16 f32_to_bf16(float f) {
    unsigned int u = __float_as_uint(f);
    unsigned int r = u + 0x7FFFu + ((u >> 16) & 1u);   // round-to-nearest-even
    return (u16)(r >> 16);
}

// ---------------- cast fp32 -> bf16 ----------------
__global__ void cast_bf16_kernel(const float* __restrict__ src,
                                 u16* __restrict__ dst, size_t n) {
    size_t i = (size_t)blockIdx.x * blockDim.x + threadIdx.x;
    size_t stride = (size_t)gridDim.x * blockDim.x;
    for (; i < n; i += stride) dst[i] = f32_to_bf16(src[i]);
}

// ---------------- init: zero out, counts, slots ----------------
__global__ void init_kernel(int* __restrict__ counts, int* __restrict__ cursors,
                            int* __restrict__ slot_token, int* __restrict__ slot_expert,
                            float* __restrict__ slot_weight, float* __restrict__ out_main) {
    size_t i = (size_t)blockIdx.x * blockDim.x + threadIdx.x;
    size_t stride = (size_t)gridDim.x * blockDim.x;
    const size_t nOut = (size_t)kT * kH;
    for (size_t j = i; j < nOut; j += stride) out_main[j] = 0.f;
    for (size_t j = i; j < (size_t)ROWS_CAP; j += stride) {
        slot_token[j]  = -1;
        slot_expert[j] = 0;
        slot_weight[j] = 0.f;
    }
    if (i < kE) { counts[i] = 0; cursors[i] = 0; }
}

// ---------------- router: one wave32 per token ----------------
__global__ void router_kernel(const float* __restrict__ x,
                              const float* __restrict__ gate_w,
                              float* __restrict__ logits_out,
                              int* __restrict__ counts,
                              int* __restrict__ top_e, float* __restrict__ top_w) {
    const int wave = threadIdx.x >> 5;
    const int lane = threadIdx.x & 31;
    const int t = blockIdx.x * 8 + wave;           // 8 waves per 256-thread block
    if (t >= kT) return;
    const float* xr = x + (size_t)t * kH;

    float acc[kE];
#pragma unroll
    for (int e = 0; e < kE; ++e) acc[e] = 0.f;
    for (int k = lane; k < kH; k += 32) {
        float xv = xr[k];
#pragma unroll
        for (int e = 0; e < kE; ++e) acc[e] += xv * gate_w[e * kH + k];
    }
#pragma unroll
    for (int off = 16; off > 0; off >>= 1) {
#pragma unroll
        for (int e = 0; e < kE; ++e) acc[e] += __shfl_xor(acc[e], off, 32);
    }
    if (lane == 0) {
        float mx = acc[0];
#pragma unroll
        for (int e = 1; e < kE; ++e) mx = fmaxf(mx, acc[e]);
        float ex[kE]; float s = 0.f;
#pragma unroll
        for (int e = 0; e < kE; ++e) { ex[e] = __expf(acc[e] - mx); s += ex[e]; }
        int i0 = 0;
#pragma unroll
        for (int e = 1; e < kE; ++e) if (acc[e] > acc[i0]) i0 = e;
        int i1 = (i0 == 0) ? 1 : 0;
#pragma unroll
        for (int e = 0; e < kE; ++e) if (e != i0 && acc[e] > acc[i1]) i1 = e;
        float p0 = ex[i0] / s, p1 = ex[i1] / s;
        float rs = p0 + p1;
#pragma unroll
        for (int e = 0; e < kE; ++e) logits_out[(size_t)t * kE + e] = acc[e];
        top_e[2 * t + 0] = i0;  top_w[2 * t + 0] = p0 / rs;
        top_e[2 * t + 1] = i1;  top_w[2 * t + 1] = p1 / rs;
        atomicAdd(&counts[i0], 1);
        atomicAdd(&counts[i1], 1);
    }
}

// ---------------- exclusive scan of counts, padded to BM (block-uniform expert) ----------------
__global__ void scan_kernel(const int* __restrict__ counts, int* __restrict__ offs) {
    if (threadIdx.x == 0 && blockIdx.x == 0) {
        int o = 0;
        for (int e = 0; e < kE; ++e) {
            offs[e] = o;
            o += ((counts[e] + BM - 1) / BM) * BM;
        }
    }
}

// ---------------- scatter tokens into expert-grouped slots ----------------
__global__ void scatter_kernel(const int* __restrict__ top_e, const float* __restrict__ top_w,
                               const int* __restrict__ offs, int* __restrict__ cursors,
                               int* __restrict__ slot_token, int* __restrict__ slot_expert,
                               float* __restrict__ slot_weight) {
    int t = blockIdx.x * blockDim.x + threadIdx.x;
    if (t >= kT) return;
#pragma unroll
    for (int j = 0; j < kTopK; ++j) {
        int e = top_e[2 * t + j];
        int pos = atomicAdd(&cursors[e], 1);
        int s = offs[e] + pos;
        slot_token[s]  = t;
        slot_expert[s] = e;
        slot_weight[s] = top_w[2 * t + j];
    }
}

// ---------------- GEMM1: g = silu(x w1^T) * (x w3^T), bf16 WMMA ----------------
// Block: 8 waves = 2 M-waves x 4 N-waves, tile 128(M) x 128(N).
// Wave tile: 64(M) x 32(N), both matrices -> 16 wmma per K-step, 16 b128 loads.
// B fragments are loaded per N-subtile just before use to keep live VGPRs ~200.
__global__ void __launch_bounds__(256, 2)
moe_gemm1(const u16* __restrict__ xb,
          const u16* __restrict__ w1b,
          const u16* __restrict__ w3b,
          const int* __restrict__ slot_token,
          const int* __restrict__ slot_expert,
          u16* __restrict__ gbuf) {
    const int wave  = threadIdx.x >> 5;
    const int lane  = threadIdx.x & 31;
    const int half  = lane >> 4;
    const int lm    = lane & 15;
    const int waveM = wave & 1;
    const int waveN = wave >> 1;
    const int mWave = blockIdx.y * BM + waveM * 64;
    const int nWave = blockIdx.x * 128 + waveN * 32;

    const int e = slot_expert[blockIdx.y * BM];   // block-uniform (segments padded to BM)
    const u16* W1 = w1b + (size_t)e * kF * kH;
    const u16* W3 = w3b + (size_t)e * kF * kH;

    const u16* arow[4];
#pragma unroll
    for (int s = 0; s < 4; ++s) {
        int tok = slot_token[mWave + s * 16 + lm];
        if (tok < 0) tok = 0;                     // padding rows: garbage, never scattered
        arow[s] = xb + (size_t)tok * kH + 8 * half;
    }

    f32x8 acc1[4][2], acc3[4][2];
#pragma unroll
    for (int s = 0; s < 4; ++s)
#pragma unroll
        for (int nc = 0; nc < 2; ++nc)
#pragma unroll
            for (int i = 0; i < 8; ++i) { acc1[s][nc][i] = 0.f; acc3[s][nc][i] = 0.f; }

    for (int k = 0; k < kH; k += 32) {
        FragU a[4];
#pragma unroll
        for (int s = 0; s < 4; ++s) {
            a[s].u[0] = *(const uint4*)(arow[s] + k);        // K = 8h .. 8h+7
            a[s].u[1] = *(const uint4*)(arow[s] + k + 16);   // K = 16+8h ..
        }
#pragma unroll
        for (int nc = 0; nc < 2; ++nc) {
            const int col = nWave + nc * 16 + lm;
            const size_t boff = (size_t)col * kH + k + 8 * half;
            FragU b1, b3;
            b1.u[0] = *(const uint4*)(W1 + boff);
            b1.u[1] = *(const uint4*)(W1 + boff + 16);
            b3.u[0] = *(const uint4*)(W3 + boff);
            b3.u[1] = *(const uint4*)(W3 + boff + 16);
#pragma unroll
            for (int s = 0; s < 4; ++s) {
                acc1[s][nc] = __builtin_amdgcn_wmma_f32_16x16x32_bf16(
                    false, a[s].v, false, b1.v, (short)0, acc1[s][nc], false, false);
                acc3[s][nc] = __builtin_amdgcn_wmma_f32_16x16x32_bf16(
                    false, a[s].v, false, b3.v, (short)0, acc3[s][nc], false, false);
            }
        }
    }

    // C layout: vgpr i -> row m = 8*half + i (within 16-row subtile), col = base + (lane&15)
#pragma unroll
    for (int s = 0; s < 4; ++s) {
#pragma unroll
        for (int nc = 0; nc < 2; ++nc) {
#pragma unroll
            for (int i = 0; i < 8; ++i) {
                float a1 = acc1[s][nc][i];
                float a3 = acc3[s][nc][i];
                float g  = (a1 / (1.f + __expf(-a1))) * a3;   // silu(a1) * a3
                int row = mWave + s * 16 + 8 * half + i;
                int col = nWave + nc * 16 + lm;
                gbuf[(size_t)row * kF + col] = f32_to_bf16(g);
            }
        }
    }
}

// ---------------- GEMM2: out[token] += weight * (g w2^T) ----------------
// Block: 8 waves = 2 M-waves x 4 N-waves, tile 128(M) x 256(N over H).
// Wave tile: 64(M) x 64(N) -> 16 wmma per K-step, 16 b128 loads.
__global__ void __launch_bounds__(256, 2)
moe_gemm2(const u16* __restrict__ gbuf,
          const u16* __restrict__ w2b,
          const int* __restrict__ slot_token,
          const int* __restrict__ slot_expert,
          const float* __restrict__ slot_weight,
          float* __restrict__ out) {
    const int wave  = threadIdx.x >> 5;
    const int lane  = threadIdx.x & 31;
    const int half  = lane >> 4;
    const int lm    = lane & 15;
    const int waveM = wave & 1;
    const int waveN = wave >> 1;
    const int mWave = blockIdx.y * BM + waveM * 64;
    const int nWave = blockIdx.x * 256 + waveN * 64;

    const int e = slot_expert[blockIdx.y * BM];
    const u16* W2 = w2b + (size_t)e * kH * kF;

    const u16* arow[4];
#pragma unroll
    for (int s = 0; s < 4; ++s)
        arow[s] = gbuf + (size_t)(mWave + s * 16 + lm) * kF + 8 * half;

    f32x8 acc[4][4];
#pragma unroll
    for (int s = 0; s < 4; ++s)
#pragma unroll
        for (int nc = 0; nc < 4; ++nc)
#pragma unroll
            for (int i = 0; i < 8; ++i) acc[s][nc][i] = 0.f;

    for (int k = 0; k < kF; k += 32) {
        FragU a[4];
#pragma unroll
        for (int s = 0; s < 4; ++s) {
            a[s].u[0] = *(const uint4*)(arow[s] + k);
            a[s].u[1] = *(const uint4*)(arow[s] + k + 16);
        }
#pragma unroll
        for (int nc = 0; nc < 4; ++nc) {
            const int col = nWave + nc * 16 + lm;
            const size_t boff = (size_t)col * kF + k + 8 * half;
            FragU b;
            b.u[0] = *(const uint4*)(W2 + boff);
            b.u[1] = *(const uint4*)(W2 + boff + 16);
#pragma unroll
            for (int s = 0; s < 4; ++s)
                acc[s][nc] = __builtin_amdgcn_wmma_f32_16x16x32_bf16(
                    false, a[s].v, false, b.v, (short)0, acc[s][nc], false, false);
        }
    }

    // epilogue: scale per-row by routing weight, scatter-add into out[token]
#pragma unroll
    for (int s = 0; s < 4; ++s) {
        int   tok8[8];
        float wgt8[8];
#pragma unroll
        for (int i = 0; i < 8; ++i) {
            int slot = mWave + s * 16 + 8 * half + i;
            tok8[i] = slot_token[slot];
            wgt8[i] = slot_weight[slot];
        }
#pragma unroll
        for (int nc = 0; nc < 4; ++nc) {
            int col = nWave + nc * 16 + lm;
#pragma unroll
            for (int i = 0; i < 8; ++i) {
                if (tok8[i] >= 0) {
                    atomicAdd(&out[(size_t)tok8[i] * kH + col], wgt8[i] * acc[s][nc][i]);
                }
            }
        }
    }
}

// ---------------- host-side launch ----------------
static inline char* ws_alloc(char*& p, size_t bytes) {
    char* r = p;
    p += (bytes + 255) & ~(size_t)255;
    return r;
}

extern "C" void kernel_launch(void* const* d_in, const int* in_sizes, int n_in,
                              void* d_out, int out_size, void* d_ws, size_t ws_size,
                              hipStream_t stream) {
    (void)in_sizes; (void)n_in; (void)out_size; (void)ws_size;

    const float* x      = (const float*)d_in[0];   // (B,S,H)
    const float* gate_w = (const float*)d_in[1];   // (E,H)
    const float* w1     = (const float*)d_in[2];   // (E,FFN,H)
    const float* w3     = (const float*)d_in[3];   // (E,FFN,H)
    const float* w2     = (const float*)d_in[4];   // (E,H,FFN)

    float* out_main = (float*)d_out;                       // T*H
    float* logits   = out_main + (size_t)kT * kH;          // T*E

    const size_t XB_N = (size_t)kT * kH;
    const size_t W1_N = (size_t)kE * kF * kH;
    const size_t W2_N = (size_t)kE * kH * kF;
    const size_t G_N  = (size_t)ROWS_CAP * kF;

    char* p = (char*)d_ws;
    u16*   xb          = (u16*)  ws_alloc(p, XB_N * sizeof(u16));
    u16*   w1b         = (u16*)  ws_alloc(p, W1_N * sizeof(u16));
    u16*   w3b         = (u16*)  ws_alloc(p, W1_N * sizeof(u16));
    u16*   w2b         = (u16*)  ws_alloc(p, W2_N * sizeof(u16));
    u16*   gbuf        = (u16*)  ws_alloc(p, G_N  * sizeof(u16));
    int*   counts      = (int*)  ws_alloc(p, kE * sizeof(int));
    int*   cursors     = (int*)  ws_alloc(p, kE * sizeof(int));
    int*   offs        = (int*)  ws_alloc(p, kE * sizeof(int));
    int*   top_e       = (int*)  ws_alloc(p, (size_t)2 * kT * sizeof(int));
    float* top_w       = (float*)ws_alloc(p, (size_t)2 * kT * sizeof(float));
    int*   slot_token  = (int*)  ws_alloc(p, (size_t)ROWS_CAP * sizeof(int));
    int*   slot_expert = (int*)  ws_alloc(p, (size_t)ROWS_CAP * sizeof(int));
    float* slot_weight = (float*)ws_alloc(p, (size_t)ROWS_CAP * sizeof(float));

    const int TB = 256;
    auto grid_for = [](size_t n, int tb) -> unsigned {
        size_t g = (n + (size_t)tb - 1) / tb;
        if (g > 4096) g = 4096;
        return (unsigned)g;
    };

    // 1) cast everything to bf16 (weights become L2-resident at 176 MB)
    cast_bf16_kernel<<<grid_for(XB_N, TB), TB, 0, stream>>>(x,  xb,  XB_N);
    cast_bf16_kernel<<<grid_for(W1_N, TB), TB, 0, stream>>>(w1, w1b, W1_N);
    cast_bf16_kernel<<<grid_for(W1_N, TB), TB, 0, stream>>>(w3, w3b, W1_N);
    cast_bf16_kernel<<<grid_for(W2_N, TB), TB, 0, stream>>>(w2, w2b, W2_N);

    // 2) init counters/slots and zero the main output
    init_kernel<<<4096, TB, 0, stream>>>(counts, cursors, slot_token, slot_expert,
                                         slot_weight, out_main);

    // 3) router (also writes router_logits output)
    router_kernel<<<kT / 8, TB, 0, stream>>>(x, gate_w, logits, counts, top_e, top_w);

    // 4) padded exclusive scan + scatter into expert-grouped slots
    scan_kernel<<<1, 32, 0, stream>>>(counts, offs);
    scatter_kernel<<<kT / TB, TB, 0, stream>>>(top_e, top_w, offs, cursors,
                                               slot_token, slot_expert, slot_weight);

    // 5) expert GEMMs (WMMA bf16)
    dim3 g1(kF / 128, ROWS_CAP / BM);   // 28 x 136
    moe_gemm1<<<g1, TB, 0, stream>>>(xb, w1b, w3b, slot_token, slot_expert, gbuf);

    dim3 g2(kH / 256, ROWS_CAP / BM);   // 4 x 136
    moe_gemm2<<<g2, TB, 0, stream>>>(gbuf, w2b, slot_token, slot_expert,
                                     slot_weight, out_main);
}
